// LSTM_18940805775841
// MI455X (gfx1250) — matrix operand order repaired and checked by reading, not compiled, not run on hardware
//
#include <hip/hip_runtime.h>
#include <stdint.h>

typedef __attribute__((ext_vector_type(16))) __bf16 v16bf;
typedef __attribute__((ext_vector_type(8)))  float  v8f;

#define Bsz 256
#define Tsz 512
#define Dsz 256
#define Hsz 1024
#define Osz 256
#define K4H 4096              // 4*H (gate-concatenated output width)
#define KTOT 1280             // D + H (fused GEMM K)

static __device__ __forceinline__ unsigned short f2bf(float f) {
  union { float f; unsigned int u; } v; v.f = f;
  unsigned int r = v.u + 0x7FFFu + ((v.u >> 16) & 1u);   // round-to-nearest-even
  return (unsigned short)(r >> 16);
}

// ---------------------------------------------------------------------------
// Pack [Wx;Wh] (4 gates) into bf16, column-major over K: WxhT[col(4096)][k(1280)]
// gate g occupies cols [g*1024, (g+1)*1024); k<256 -> W_*x, k>=256 -> W_*h.
// ---------------------------------------------------------------------------
__global__ void pack_wxh(const float* __restrict__ Wgx, const float* __restrict__ Wgh,
                         const float* __restrict__ Wix, const float* __restrict__ Wih,
                         const float* __restrict__ Wfx, const float* __restrict__ Wfh,
                         const float* __restrict__ Wox, const float* __restrict__ Woh,
                         unsigned short* __restrict__ WxhT) {
  int idx = blockIdx.x * blockDim.x + threadIdx.x;
  if (idx >= K4H * KTOT) return;
  int col = idx / KTOT;
  int k   = idx % KTOT;
  int g = col >> 10;
  int j = col & 1023;
  const float* Wx[4] = {Wgx, Wix, Wfx, Wox};
  const float* Wh[4] = {Wgh, Wih, Wfh, Woh};
  float v = (k < Dsz) ? Wx[g][k * Hsz + j] : Wh[g][(k - Dsz) * Hsz + j];
  WxhT[idx] = f2bf(v);
}

// Pack W_ph [H,O] -> bf16 column-major: WpT[o(256)][k(1024)]
__global__ void pack_wp(const float* __restrict__ Wph, unsigned short* __restrict__ WpT) {
  int idx = blockIdx.x * blockDim.x + threadIdx.x;
  if (idx >= Osz * Hsz) return;
  int o = idx / Hsz, k = idx % Hsz;
  WpT[idx] = f2bf(Wph[k * Osz + o]);
}

// ---------------------------------------------------------------------------
// One LSTM time step: z = [x_t, h_prev] @ WxhT^T, then gate update.
// WG: 256 thr = 8 waves; tile = 64 batch rows x 64 H-cols x 4 gates.
// wave w: mw=w&1 -> 32-row half (2 M-tiles), nq2=w>>1 -> 16-col tile.
// acc[mt][gate]: 8 WMMAs per K-slab; each B fragment reused by 2 WMMAs.
// grid = (H/64=16, B/64=4).
// ---------------------------------------------------------------------------
__global__ __launch_bounds__(256)
void lstm_step(const float* __restrict__ x,               // [B,T,D] fp32
               const unsigned short* __restrict__ WxhT,   // [4096][1280] bf16
               const unsigned short* __restrict__ h_prev, // [B][H] bf16
               unsigned short* __restrict__ h_next,       // [B][H] bf16
               float* __restrict__ c,                     // [B][H] fp32
               const float* __restrict__ bg, const float* __restrict__ bi,
               const float* __restrict__ bf_, const float* __restrict__ bo,
               int t) {
  __shared__ __align__(16) unsigned short As[64 * 40];    // A tile, padded stride

  const int tid  = threadIdx.x;
  const int lane = tid & 31;
  const int w    = tid >> 5;
  const int mw   = w & 1;          // 32-row half
  const int nq2  = w >> 1;         // 16-col tile
  const int ln   = lane & 15;
  const int hh   = lane >> 4;      // lane half (K split)

  const int cg  = blockIdx.x;           // H-column group (64 cols)
  const int bm0 = blockIdx.y * 64;      // batch row base

  v8f acc[2][4] = {};                   // [M-tile][gate]

  const int srow  = tid >> 2;           // 0..63 staging row
  const int skoff = (tid & 3) * 8;      // 0,8,16,24 staging k offset (8 bf16)

  for (int kb = 0; kb < KTOT; kb += 32) {
    // ---- stage A[64 rows][32 k] bf16 into LDS ----
    {
      int b = bm0 + srow;
      unsigned short tmp[8];
      if (kb < Dsz) {
        const float* src = x + ((size_t)b * Tsz + t) * Dsz + kb + skoff;
        const float4 f0 = ((const float4*)src)[0];
        const float4 f1 = ((const float4*)src)[1];
        tmp[0] = f2bf(f0.x); tmp[1] = f2bf(f0.y); tmp[2] = f2bf(f0.z); tmp[3] = f2bf(f0.w);
        tmp[4] = f2bf(f1.x); tmp[5] = f2bf(f1.y); tmp[6] = f2bf(f1.z); tmp[7] = f2bf(f1.w);
      } else {
        *(uint4*)tmp = *(const uint4*)(h_prev + (size_t)b * Hsz + (kb - Dsz) + skoff);
      }
      *(uint4*)&As[srow * 40 + skoff] = *(uint4*)tmp;
    }
    __syncthreads();

    // ---- A fragments (2 x 16x32) from LDS per ISA layout ----
    union { v16bf v; uint4 q[2]; } afrag[2];
    #pragma unroll
    for (int mt = 0; mt < 2; ++mt) {
      int row = mw * 32 + mt * 16 + ln;
      afrag[mt].q[0] = *(const uint4*)&As[row * 40 + hh * 8];        // K = 8h+0..7
      afrag[mt].q[1] = *(const uint4*)&As[row * 40 + 16 + hh * 8];   // K = 16+8h+0..7
    }

    // ---- B fragments (32x16) from L2-resident WxhT; 4 gates, reused 2x ----
    #pragma unroll
    for (int g = 0; g < 4; ++g) {
      union { v16bf v; uint4 q[2]; } bfrag;
      int colg = g * Hsz + cg * 64 + nq2 * 16 + ln;
      const uint4* p = (const uint4*)(WxhT + (size_t)colg * KTOT + kb);
      bfrag.q[0] = p[hh];
      bfrag.q[1] = p[2 + hh];
      acc[0][g] = __builtin_amdgcn_wmma_f32_16x16x32_bf16(
          false, afrag[0].v, false, bfrag.v, (short)0, acc[0][g], false, false);
      acc[1][g] = __builtin_amdgcn_wmma_f32_16x16x32_bf16(
          false, afrag[1].v, false, bfrag.v, (short)0, acc[1][g], false, false);
    }
    __syncthreads();
  }

  // ---- gate nonlinearity + state update, all lane-local (C layout) ----
  const int colH = cg * 64 + nq2 * 16 + ln;
  const float vbg = bg[colH], vbi = bi[colH], vbf = bf_[colH], vbo = bo[colH];
  #pragma unroll
  for (int mt = 0; mt < 2; ++mt) {
    #pragma unroll
    for (int vr = 0; vr < 8; ++vr) {
      int rowB = bm0 + mw * 32 + mt * 16 + hh * 8 + vr;
      size_t gidx = (size_t)rowB * Hsz + colH;
      float gv = tanhf(acc[mt][0][vr] + vbg);
      float iv = 1.f / (1.f + expf(-(acc[mt][1][vr] + vbi)));
      float fv = 1.f / (1.f + expf(-(acc[mt][2][vr] + vbf)));
      float ov = 1.f / (1.f + expf(-(acc[mt][3][vr] + vbo)));
      float cn = gv * iv + c[gidx] * fv;
      c[gidx] = cn;
      h_next[gidx] = f2bf(tanhf(cn) * ov);
    }
  }
}

// ---------------------------------------------------------------------------
// logits = h @ W_ph  (bias + softmax applied in softmax_rows)
// grid = (O/64=4, B/32=8); wave: one 16x16 tile.
// ---------------------------------------------------------------------------
__global__ __launch_bounds__(256)
void proj_gemm(const unsigned short* __restrict__ h,    // [B][H] bf16
               const unsigned short* __restrict__ WpT,  // [O][H] bf16
               float* __restrict__ logits) {            // [B][O] fp32
  __shared__ __align__(16) unsigned short As[32 * 40];
  const int tid = threadIdx.x, lane = tid & 31, w = tid >> 5;
  const int mq = w & 1, nq2 = w >> 1;
  const int ln = lane & 15, hh = lane >> 4;
  const int cg = blockIdx.x, bm0 = blockIdx.y * 32;
  const int srow = tid >> 3, skoff = (tid & 7) * 4;

  v8f acc = {};
  for (int kb = 0; kb < Hsz; kb += 32) {
    *(uint2*)&As[srow * 40 + skoff] =
        *(const uint2*)(h + (size_t)(bm0 + srow) * Hsz + kb + skoff);
    __syncthreads();

    union { v16bf v; uint4 q[2]; } afrag, bfrag;
    int row = mq * 16 + ln;
    afrag.q[0] = *(const uint4*)&As[row * 40 + hh * 8];
    afrag.q[1] = *(const uint4*)&As[row * 40 + 16 + hh * 8];
    int o = cg * 64 + nq2 * 16 + ln;
    const uint4* p = (const uint4*)(WpT + (size_t)o * Hsz + kb);
    bfrag.q[0] = p[hh];
    bfrag.q[1] = p[2 + hh];
    acc = __builtin_amdgcn_wmma_f32_16x16x32_bf16(
        false, afrag.v, false, bfrag.v, (short)0, acc, false, false);
    __syncthreads();
  }
  #pragma unroll
  for (int vr = 0; vr < 8; ++vr) {
    int rowB = bm0 + mq * 16 + hh * 8 + vr;
    int col  = cg * 64 + nq2 * 16 + ln;
    logits[(size_t)rowB * Osz + col] = acc[vr];
  }
}

// softmax over axis=1 (O=256), one block per row, + bias b_p
__global__ __launch_bounds__(256)
void softmax_rows(const float* __restrict__ logits, const float* __restrict__ bp,
                  float* __restrict__ out) {
  __shared__ float red[256];
  int row = blockIdx.x, tid = threadIdx.x;
  float v = logits[(size_t)row * Osz + tid] + bp[tid];
  red[tid] = v; __syncthreads();
  for (int s = 128; s > 0; s >>= 1) {
    if (tid < s) red[tid] = fmaxf(red[tid], red[tid + s]);
    __syncthreads();
  }
  float m = red[0]; __syncthreads();
  float e = expf(v - m);
  red[tid] = e; __syncthreads();
  for (int s = 128; s > 0; s >>= 1) {
    if (tid < s) red[tid] += red[tid + s];
    __syncthreads();
  }
  out[(size_t)row * Osz + tid] = e / red[0];
}

// ---------------------------------------------------------------------------
extern "C" void kernel_launch(void* const* d_in, const int* in_sizes, int n_in,
                              void* d_out, int out_size, void* d_ws, size_t ws_size,
                              hipStream_t stream) {
  const float* x   = (const float*)d_in[0];
  const float* Wgx = (const float*)d_in[1];
  const float* Wgh = (const float*)d_in[2];
  const float* bg  = (const float*)d_in[3];
  const float* Wix = (const float*)d_in[4];
  const float* Wih = (const float*)d_in[5];
  const float* bi  = (const float*)d_in[6];
  const float* Wfx = (const float*)d_in[7];
  const float* Wfh = (const float*)d_in[8];
  const float* bf  = (const float*)d_in[9];
  const float* Wox = (const float*)d_in[10];
  const float* Woh = (const float*)d_in[11];
  const float* bo  = (const float*)d_in[12];
  const float* Wph = (const float*)d_in[13];
  const float* bp  = (const float*)d_in[14];

  uint8_t* ws = (uint8_t*)d_ws;
  size_t off = 0;
  unsigned short* WxhT = (unsigned short*)(ws + off); off += (size_t)K4H * KTOT * 2; // 10.0 MB
  unsigned short* WpT  = (unsigned short*)(ws + off); off += (size_t)Osz * Hsz * 2;  // 0.5 MB
  unsigned short* h0   = (unsigned short*)(ws + off); off += (size_t)Bsz * Hsz * 2;  // 0.5 MB
  unsigned short* h1   = (unsigned short*)(ws + off); off += (size_t)Bsz * Hsz * 2;  // 0.5 MB
  float* cbuf          = (float*)(ws + off);          off += (size_t)Bsz * Hsz * 4;  // 1.0 MB
  float* logits        = (float*)(ws + off);          off += (size_t)Bsz * Osz * 4;  // 0.25 MB

  // zero initial h and c every call (ws is poisoned; not re-poisoned between replays)
  hipMemsetAsync(h0,   0, (size_t)Bsz * Hsz * 2, stream);
  hipMemsetAsync(cbuf, 0, (size_t)Bsz * Hsz * 4, stream);

  pack_wxh<<<(K4H * KTOT + 255) / 256, 256, 0, stream>>>(Wgx, Wgh, Wix, Wih,
                                                         Wfx, Wfh, Wox, Woh, WxhT);
  pack_wp<<<(Osz * Hsz + 255) / 256, 256, 0, stream>>>(Wph, WpT);

  for (int t = 0; t < Tsz; ++t) {
    const unsigned short* hp = (t & 1) ? h1 : h0;
    unsigned short*       hn = (t & 1) ? h0 : h1;
    lstm_step<<<dim3(Hsz / 64, Bsz / 64), 256, 0, stream>>>(
        x, WxhT, hp, hn, cbuf, bg, bi, bf, bo, t);
  }
  // T=512 even -> final h is in h0
  proj_gemm<<<dim3(Osz / 64, Bsz / 32), 256, 0, stream>>>(h0, WpT, logits);
  softmax_rows<<<Bsz, 256, 0, stream>>>(logits, bp, (float*)d_out);
}
